// RNN_52836687675754
// MI455X (gfx1250) — compile-verified, matrix operand order checked by reading, not run on hardware
//
#include <hip/hip_runtime.h>

// Problem constants (match reference)
#define H  1024
#define I  256
#define O  256
#define BN 64   // batch
#define T  512

#define NWG   64   // workgroups in persistent recurrent kernel (must be co-resident)
#define BLK   128  // 4 waves of 32

typedef __attribute__((ext_vector_type(16))) __bf16 v16bf;
typedef __attribute__((ext_vector_type(8)))  __bf16 v8bf;
typedef __attribute__((ext_vector_type(8)))  float  v8f;

// ---------- helpers ----------

__device__ __forceinline__ unsigned short f32_to_bf16_bits(float f) {
    unsigned u = __builtin_bit_cast(unsigned, f);
    // round-to-nearest-even
    return (unsigned short)((u + 0x7FFFu + ((u >> 16) & 1u)) >> 16);
}

// A/B fragment for v_wmma_f32_16x16x32_bf16.
// 16-bit A 16x32 layout (ISA 7.12.2): lane L holds row M=L%16;
// lanes 0-15: K = kb+0..7 and kb+16..23 ; lanes 16-31: K = kb+8..15 and kb+24..31.
// B (32x16) is the mirror image with "row" = column N, which for [out,in]
// row-major weights is exactly the same addressing pattern.
__device__ __forceinline__ v16bf load_frag(const unsigned short* __restrict__ p,
                                           size_t row0, int ld, int kb, int lane) {
    int rr = lane & 15;
    int ko = kb + ((lane & 16) >> 1);           // +8 for upper half-wave
    const unsigned short* q = p + (row0 + (size_t)rr) * (size_t)ld + ko;
    v8bf lo = *reinterpret_cast<const v8bf*>(q);       // 16B: K group 0
    v8bf hi = *reinterpret_cast<const v8bf*>(q + 16);  // 16B: K group 1
    return __builtin_shufflevector(lo, hi, 0,1,2,3,4,5,6,7,8,9,10,11,12,13,14,15);
}

__device__ __forceinline__ v8f wmma_bf16(v16bf a, v16bf b, v8f c) {
    return __builtin_amdgcn_wmma_f32_16x16x32_bf16(
        /*neg_a=*/false, a, /*neg_b=*/false, b,
        /*c_mod=*/(short)0, c, /*reuse_a=*/false, /*reuse_b=*/false);
}

// ---------- kernel 1: f32 -> bf16 convert ----------

__global__ void __launch_bounds__(256)
convert_bf16(const float* __restrict__ src, unsigned short* __restrict__ dst, int n) {
    int i = blockIdx.x * 256 + threadIdx.x;
    if (i < n) dst[i] = f32_to_bf16_bits(src[i]);
}

// ---------- kernel 2: x_proj = inputs @ w_in^T + b_in, stored time-major [T][BN][H] ----------

__global__ void __launch_bounds__(BLK)
xproj_gemm(const unsigned short* __restrict__ in_bf,   // [BN*T][I] bf16 (rows = b*T+t)
           const unsigned short* __restrict__ win_bf,  // [H][I] bf16
           const float* __restrict__ b_in,
           float* __restrict__ xproj) {                // [T][BN][H] f32
    int wave = threadIdx.x >> 5;
    int lane = threadIdx.x & 31;
    int tile = blockIdx.x * 4 + wave;       // (BN*T/16)*(H/16) tiles
    int mtile = tile >> 6;                  // H/16 == 64 ntiles
    int ntile = tile & 63;
    size_t m0 = (size_t)mtile * 16;
    size_t n0 = (size_t)ntile * 16;

    // Two independent accumulator chains (even/odd K blocks) so consecutive
    // WMMAs are not serially dependent in the matrix pipe.
    v8f acc0 = {}, acc1 = {};
#pragma unroll
    for (int kb = 0; kb < I; kb += 64) {
        acc0 = wmma_bf16(load_frag(in_bf,  m0, I, kb,      lane),
                         load_frag(win_bf, n0, I, kb,      lane), acc0);
        acc1 = wmma_bf16(load_frag(in_bf,  m0, I, kb + 32, lane),
                         load_frag(win_bf, n0, I, kb + 32, lane), acc1);
    }
    int ng = (int)n0 + (lane & 15);
    int mb = (int)m0 + ((lane & 16) >> 1);
    float bi = b_in[ng];
#pragma unroll
    for (int r = 0; r < 8; ++r) {
        int m = mb + r;                 // flat row = b*T + t
        int b = m >> 9;                 // /T
        int t = m & (T - 1);
        xproj[((size_t)t * BN + b) * H + ng] = (acc0[r] + acc1[r]) + bi;
    }
}

// ---------- kernel 3: persistent recurrent scan ----------

__global__ void __launch_bounds__(BLK)
rnn_scan(const unsigned short* __restrict__ wrec_bf,  // [H][H] bf16
         const unsigned short* __restrict__ wout_bf,  // [O][H] bf16
         const float* __restrict__ xproj,             // [T][BN][H]
         const float* __restrict__ b_rec,
         const float* __restrict__ b_out,
         const float* __restrict__ scale,
         const float* __restrict__ shift,
         unsigned short* __restrict__ hbuf,           // 2 * [BN][H] bf16 (double buffer)
         unsigned* __restrict__ bars,                 // [T] zeroed counters
         float* __restrict__ outs,                    // [BN][T][O]
         float* __restrict__ hids) {                  // [BN][T][H]
    const int wg   = blockIdx.x;          // 0..NWG-1
    const int wave = threadIdx.x >> 5;    // 0..3
    const int lane = threadIdx.x & 31;

    // Stage this WG's weight slices into LDS once; rows padded to 129 uint4
    // (1032 bf16) so 16 rows hit distinct bank groups for ds_load_b128.
    __shared__ uint4 lds_w[2 * 16 * 129];
    __shared__ float red[3][32][8];       // readout K-partials from waves 1..3
    {
        const uint4* wr = reinterpret_cast<const uint4*>(wrec_bf + (size_t)wg * 16 * H);
        const uint4* wo = reinterpret_cast<const uint4*>(wout_bf + (size_t)(wg & 15) * 16 * H);
        for (int idx = threadIdx.x; idx < 16 * 128; idx += BLK) {
            int row = idx >> 7, col = idx & 127;
            lds_w[row * 129 + col]            = wr[idx];
            lds_w[16 * 129 + row * 129 + col] = wo[idx];
        }
    }
    __syncthreads();
    const unsigned short* lds_wrec = reinterpret_cast<const unsigned short*>(lds_w);
    const unsigned short* lds_wout = lds_wrec + 16 * 1032;

    const int rec_n0 = wg * 16;           // this WG's h_new column slice
    const int rec_m0 = wave * 16;         // this wave's batch-row tile
    const int out_o0 = (wg & 15) * 16;    // readout tile owned by this WG
    const int out_m0 = (wg >> 4) * 16;
    const int out_k0 = wave * (H / 4);    // readout K-range per wave

    const int nloc  = lane & 15;
    const int mhalf = (lane & 16) >> 1;   // 0 or 8

    const float brec_n = b_rec[rec_n0 + nloc];
    const float bo = b_out[out_o0 + nloc];
    const float sc = scale[out_o0 + nloc];
    const float sh = shift[out_o0 + nloc];

    for (int t = 0; t < T; ++t) {
        const unsigned short* hin  = hbuf + (size_t)(t & 1) * (BN * H);
        unsigned short*       hout = hbuf + (size_t)((t + 1) & 1) * (BN * H);

        // ---- h_new tile: [16x16] over K=H, A from L2-resident h, B from LDS.
        //      Two independent WMMA chains to hide matrix-pipe latency (this
        //      kernel runs ~1 wave/SIMD, so ILP must come from within the wave).
        v8f acc0 = {}, acc1 = {};
#pragma unroll 8
        for (int kb = 0; kb < H; kb += 64) {
            acc0 = wmma_bf16(load_frag(hin, (size_t)rec_m0, H, kb, lane),
                             load_frag(lds_wrec, 0, 1032, kb, lane), acc0);
            acc1 = wmma_bf16(load_frag(hin, (size_t)rec_m0, H, kb + 32, lane),
                             load_frag(lds_wrec, 0, 1032, kb + 32, lane), acc1);
        }

        const int ng = rec_n0 + nloc;
        const float* xp = xproj + ((size_t)t * BN) * H;
#pragma unroll
        for (int r = 0; r < 8; ++r) {
            int mg = rec_m0 + mhalf + r;                 // batch row
            float v = (acc0[r] + acc1[r]) + brec_n + xp[(size_t)mg * H + ng];
            v = v > 0.f ? v : 0.f;
            hids[((size_t)mg * T + t) * H + ng] = v;     // f32 hidden output
            hout[(size_t)mg * H + ng] = f32_to_bf16_bits(v);
            if (t + 1 < T)                               // pull next step's x_proj
                __builtin_prefetch(xproj + ((size_t)(t + 1) * BN + mg) * H + ng, 0, 1);
        }

        // ---- device-wide barrier (release h_new, acquire everyone's slices) ----
        __threadfence();                 // release own stores to device scope
        __syncthreads();
        if (threadIdx.x == 0) {
            unsigned* c = bars + t;
            atomicAdd(c, 1u);
            while (__hip_atomic_load(c, __ATOMIC_RELAXED, __HIP_MEMORY_SCOPE_AGENT)
                   < (unsigned)NWG)
                __builtin_amdgcn_s_sleep(1);
        }
        __syncthreads();
        __threadfence();                 // acquire

        // ---- readout for this step: K split across the 4 waves (8 WMMAs each),
        //      f32 partials combined through LDS, wave 0 finishes the epilogue.
        //      (Balances the per-step WMMA chain: 32+8 per wave instead of 32/64.)
        v8f oc0 = {}, oc1 = {};
#pragma unroll
        for (int kk = 0; kk < H / 4; kk += 64) {
            int kb = out_k0 + kk;
            oc0 = wmma_bf16(load_frag(hout, (size_t)out_m0, H, kb, lane),
                            load_frag(lds_wout, 0, 1032, kb, lane), oc0);
            oc1 = wmma_bf16(load_frag(hout, (size_t)out_m0, H, kb + 32, lane),
                            load_frag(lds_wout, 0, 1032, kb + 32, lane), oc1);
        }
        if (wave != 0) {
#pragma unroll
            for (int r = 0; r < 8; ++r) red[wave - 1][lane][r] = oc0[r] + oc1[r];
        }
        __syncthreads();
        // WAR on `red` is safe: waves can't reach their next `red` store without
        // passing barrier t+1, which wave 0 only joins after reading `red` here.
        if (wave == 0) {
            const int og = out_o0 + nloc;
#pragma unroll
            for (int r = 0; r < 8; ++r) {
                float o = (oc0[r] + oc1[r])
                        + red[0][lane][r] + red[1][lane][r] + red[2][lane][r];
                int mg = out_m0 + mhalf + r;
                outs[((size_t)mg * T + t) * O + og] = sc * (o + bo) - sh;
            }
        }
    }
}

// ---------- host ----------

extern "C" void kernel_launch(void* const* d_in, const int* in_sizes, int n_in,
                              void* d_out, int out_size, void* d_ws, size_t ws_size,
                              hipStream_t stream) {
    (void)in_sizes; (void)n_in; (void)out_size; (void)ws_size;

    const float* inputs = (const float*)d_in[0];   // [BN,T,I]
    const float* w_rec  = (const float*)d_in[1];   // [H,H]
    const float* b_rec  = (const float*)d_in[2];
    const float* w_in   = (const float*)d_in[3];   // [H,I]
    const float* b_in   = (const float*)d_in[4];
    const float* w_out  = (const float*)d_in[5];   // [O,H]
    const float* b_out  = (const float*)d_in[6];
    const float* scale  = (const float*)d_in[7];
    const float* shift  = (const float*)d_in[8];

    char* ws = (char*)d_ws;
    size_t off = 0;
    unsigned short* wrec_bf = (unsigned short*)(ws + off); off += (size_t)H * H * 2;
    unsigned short* win_bf  = (unsigned short*)(ws + off); off += (size_t)H * I * 2;
    unsigned short* wout_bf = (unsigned short*)(ws + off); off += (size_t)O * H * 2;
    unsigned short* in_bf   = (unsigned short*)(ws + off); off += (size_t)BN * T * I * 2;
    float*          xproj   = (float*)(ws + off);          off += (size_t)T * BN * H * 4;
    unsigned short* hbuf    = (unsigned short*)(ws + off); off += (size_t)2 * BN * H * 2;
    unsigned*       bars    = (unsigned*)(ws + off);       off += (size_t)T * 4;

    float* outs = (float*)d_out;
    float* hids = outs + (size_t)BN * T * O;

    // zero h0 (double buffer) + barrier counters — hbuf/bars are contiguous
    hipMemsetAsync(hbuf, 0, (size_t)2 * BN * H * 2 + (size_t)T * 4, stream);

    auto cvt = [&](const float* s, unsigned short* d, int n) {
        convert_bf16<<<(n + 255) / 256, 256, 0, stream>>>(s, d, n);
    };
    cvt(w_rec,  wrec_bf, H * H);
    cvt(w_in,   win_bf,  H * I);
    cvt(w_out,  wout_bf, O * H);
    cvt(inputs, in_bf,   BN * T * I);

    // (BN*T/16)*(H/16) = 131072 wave-tiles, 4 per block
    xproj_gemm<<<(BN * T / 16) * (H / 16) / 4, BLK, 0, stream>>>(in_bf, win_bf, b_in, xproj);

    rnn_scan<<<NWG, BLK, 0, stream>>>(wrec_bf, wout_bf, xproj,
                                      b_rec, b_out, scale, shift,
                                      hbuf, bars, outs, hids);
}
